// CausalSelfAttention_34179349741605
// MI455X (gfx1250) — compile-verified, matrix operand order checked by reading, not run on hardware
//
#include <hip/hip_runtime.h>
#include <hip/hip_bf16.h>

// ---------------------------------------------------------------------------
// CausalSelfAttention for MI455X (gfx1250, wave32, WMMA, async-to-LDS staging)
// B=2, T=2048, C=1024, H=16, D=64
// ---------------------------------------------------------------------------

#define DEVI __device__ __forceinline__

typedef __attribute__((ext_vector_type(16))) __bf16        v16bf;
typedef __attribute__((ext_vector_type(8)))  float         v8f;
typedef __attribute__((ext_vector_type(4)))  unsigned int  u32x4;

union Frag { u32x4 u4[2]; v16bf v; };

static constexpr int Bn = 2;
static constexpr int T  = 2048;
static constexpr int C  = 1024;
static constexpr int H  = 16;
static constexpr int D  = 64;
static constexpr int C3 = 3 * C;

// float -> bf16 round-to-nearest-even
DEVI unsigned short f2bf(float x) {
    unsigned u = __builtin_bit_cast(unsigned, x);
    unsigned r = u + 0x7fffu + ((u >> 16) & 1u);
    return (unsigned short)(r >> 16);
}
DEVI float bf2f(unsigned short s) {
    return __builtin_bit_cast(float, (unsigned)s << 16);
}

DEVI v8f bf16_wmma(v16bf a, v16bf b, v8f c) {
    // v_wmma_f32_16x16x32_bf16  (8 args: neg_a, A, neg_b, B, c_mod, C, reuse_a, reuse_b)
    return __builtin_amdgcn_wmma_f32_16x16x32_bf16(false, a, false, b, (short)0, c, false, false);
}

// Load a 16x32 bf16 A-style fragment from a row-major [rows, ld] bf16 matrix.
// ISA layout: lane L holds row (rowBase + L%16); VGPRs 0-3 hold K = 8*(L/16)+0..7,
// VGPRs 4-7 hold K = 16 + 8*(L/16)+0..7 -> two contiguous 16-byte loads.
// (B fragments are loaded identically from the transposed matrix.)
DEVI v16bf load_frag(const unsigned short* base, int rowBase, int ld, int kBase, int lane) {
    Frag f;
    const unsigned short* p = base + (size_t)(rowBase + (lane & 15)) * ld + kBase + ((lane >> 4) << 3);
    f.u4[0] = *(const u32x4*)(p);
    f.u4[1] = *(const u32x4*)(p + 16);
    return f.v;
}

// B fragment out of an LDS-staged 64x32 bf16 tile (row-major, ld = 32).
DEVI v16bf lds_fragB(const unsigned short* sB, int f, int lane) {
    Frag fr;
    const unsigned short* p = sB + (f * 16 + (lane & 15)) * 32 + ((lane >> 4) << 3);
    fr.u4[0] = *(const u32x4*)(p);
    fr.u4[1] = *(const u32x4*)(p + 16);
    return fr.v;
}

// Async-copy one 16-byte segment of the 64x32 B tile into LDS (one b128 per
// thread per K-chunk; 256 threads cover 4 KB).  Tracked by ASYNCcnt.
#define ISSUE_B_TILE(buf, kk)                                                         \
    do {                                                                              \
        const unsigned short* _g = Bt + (size_t)(Nbase + row) * C + (kk) + seg;       \
        unsigned _lds = (unsigned)(size_t)(&sB[buf][row * 32 + seg]);                 \
        asm volatile("global_load_async_to_lds_b128 %0, %1, off"                      \
                     :: "v"(_lds), "v"(_g) : "memory");                               \
    } while (0)

#define WAIT_ASYNC() asm volatile("s_wait_asynccnt 0x0" ::: "memory")

// ---------------------------------------------------------------------------
// Kernel 1: flat cast f32 -> bf16
// ---------------------------------------------------------------------------
__global__ void __launch_bounds__(256) cast_bf16_kernel(const float* __restrict__ src,
                                                        unsigned short* __restrict__ dst, int n) {
    int i = blockIdx.x * 256 + threadIdx.x;
    if (i < n) dst[i] = f2bf(src[i]);
}

// ---------------------------------------------------------------------------
// Kernel 2: transpose-cast: src [K, N] f32 (row-major) -> dst [N, K] bf16
// ---------------------------------------------------------------------------
__global__ void __launch_bounds__(256) transpose_cast_kernel(const float* __restrict__ src,
                                                             unsigned short* __restrict__ dst,
                                                             int K, int N) {
    int idx = blockIdx.x * 256 + threadIdx.x;   // idx = c*K + k  (coalesced writes)
    if (idx >= K * N) return;
    int k = idx % K;
    int c = idx / K;
    dst[idx] = f2bf(src[(size_t)k * N + c]);
}

// ---------------------------------------------------------------------------
// Kernel 3: QKV GEMM.  qkv = x @ W_attn + b_attn
//   A  = x  bf16 [4096, 1024] row-major
//   Bt = W_attn^T bf16 [3072, 1024]
// Block = 8 waves = 128 rows x 64 cols.  The shared 64x32 B tile is staged
// into LDS with double-buffered GLOBAL_LOAD_ASYNC_TO_LDS_B128 (ASYNCcnt),
// overlapping the next chunk's fetch with the current chunk's WMMAs.
// Epilogue scatters into q,k [B,H,T,D] and vT [B,H,D,T] (bf16).
// ---------------------------------------------------------------------------
__global__ void __launch_bounds__(256) qkv_gemm_kernel(const unsigned short* __restrict__ xb,
                                                       const unsigned short* __restrict__ Bt,
                                                       const float* __restrict__ b_attn,
                                                       unsigned short* __restrict__ qo,
                                                       unsigned short* __restrict__ ko,
                                                       unsigned short* __restrict__ vTo) {
    __shared__ unsigned short sB[2][64 * 32];

    const int tid  = threadIdx.x;
    const int lane = tid & 31;
    const int wid  = tid >> 5;
    const int Mbase = blockIdx.x * 128 + wid * 16;
    const int Nbase = blockIdx.y * 64;
    const int row = tid >> 2;            // 0..63  (N row of B tile)
    const int seg = (tid & 3) << 3;      // 0,8,16,24 (K segment, 8 bf16 = 16B)

    v8f acc[4] = {{}, {}, {}, {}};
    ISSUE_B_TILE(0, 0);
    for (int i = 0; i < 32; ++i) {
        const int kk = i << 5;
        WAIT_ASYNC();
        __syncthreads();                              // staged tile visible to all waves
        if (i + 1 < 32) ISSUE_B_TILE((i + 1) & 1, kk + 32);

        v16bf a = load_frag(xb, Mbase, C, kk, lane);
        const unsigned short* bs = sB[i & 1];
        #pragma unroll
        for (int f = 0; f < 4; ++f)
            acc[f] = bf16_wmma(a, lds_fragB(bs, f, lane), acc[f]);
    }

    const int hh = lane >> 4;
    const int n  = lane & 15;
    #pragma unroll
    for (int f = 0; f < 4; ++f) {
        const int c     = Nbase + f * 16 + n;
        const int which = c >> 10;            // 0=q 1=k 2=v
        const int cm    = c & 1023;
        const int head  = cm >> 6;
        const int d     = cm & 63;
        const float bias = b_attn[c];
        #pragma unroll
        for (int i = 0; i < 8; ++i) {
            const int r  = Mbase + i + 8 * hh;   // row in [B*T]
            const int bb = r >> 11;              // T = 2048
            const int t  = r & 2047;
            const unsigned short bv = f2bf(acc[f][i] + bias);
            if (which == 0)
                qo[((((size_t)bb * H + head) * T + t) << 6) + d] = bv;
            else if (which == 1)
                ko[((((size_t)bb * H + head) * T + t) << 6) + d] = bv;
            else
                vTo[(((size_t)bb * H + head) * D + d) * T + t] = bv;
        }
    }
}

// ---------------------------------------------------------------------------
// Kernel 4: RoPE applied in place to q and k  [B,H,T,D] bf16 (rotate_half).
// Thread per (b,h,t,d<32): pair (d, d+32), angle = t * 10000^(-d/32).
// ---------------------------------------------------------------------------
__global__ void __launch_bounds__(256) rope_kernel(unsigned short* __restrict__ qb,
                                                   unsigned short* __restrict__ kb) {
    const int idx = blockIdx.x * 256 + threadIdx.x;  // [B*H*T*32) = 2^21
    const int d   = idx & 31;
    const int t   = (idx >> 5) & 2047;
    const int bh  = idx >> 16;
    const size_t base = ((size_t)bh * T + t) * D;

    const float ang = (float)t * __powf(10000.0f, -(float)d * (1.0f / 32.0f));
    float s, c;
    __sincosf(ang, &s, &c);

    float q1 = bf2f(qb[base + d]), q2 = bf2f(qb[base + d + 32]);
    qb[base + d]      = f2bf(q1 * c - q2 * s);
    qb[base + d + 32] = f2bf(q2 * c + q1 * s);

    float k1 = bf2f(kb[base + d]), k2 = bf2f(kb[base + d + 32]);
    kb[base + d]      = f2bf(k1 * c - k2 * s);
    kb[base + d + 32] = f2bf(k2 * c + k1 * s);
}

// ---------------------------------------------------------------------------
// Kernel 5: causal flash attention.  One wave per (b, h, 16-query tile).
// Online softmax over 32-key chunks.  S via 2 WMMAs per 16-key subtile,
// P staged in LDS (16x32 bf16, per-wave private) and reloaded as an A
// fragment, PV via 4 WMMAs.  Output y [B,T,C] bf16.
// ---------------------------------------------------------------------------
__global__ void __launch_bounds__(256) attn_kernel(const unsigned short* __restrict__ qb,
                                                   const unsigned short* __restrict__ kb,
                                                   const unsigned short* __restrict__ vT,
                                                   unsigned short* __restrict__ yb) {
    __shared__ unsigned short pstage[8][16 * 32];

    const int lane = threadIdx.x & 31;
    const int wid  = threadIdx.x >> 5;
    const int gw   = blockIdx.x * 8 + wid;      // [0, 4096)
    const int qt   = gw & 127;                  // T/16 = 128 tiles
    const int h    = (gw >> 7) & 15;
    const int b    = gw >> 11;
    const int qbase = qt * 16;

    const unsigned short* qp = qb + ((size_t)b * H + h) * T * D;
    const unsigned short* kp = kb + ((size_t)b * H + h) * T * D;
    const unsigned short* vp = vT + ((size_t)b * H + h) * D * T;

    const v16bf qa0 = load_frag(qp, qbase, D, 0,  lane);
    const v16bf qa1 = load_frag(qp, qbase, D, 32, lane);

    v8f o[4] = {{}, {}, {}, {}};
    float mrow[8], lrow[8];
    #pragma unroll
    for (int i = 0; i < 8; ++i) { mrow[i] = -3.0e38f; lrow[i] = 0.0f; }

    const int hh = lane >> 4;
    const int n  = lane & 15;
    const float scale = 0.125f;                 // 1/sqrt(64)
    const int kend = qbase + 16;

    for (int kc = 0; kc < kend; kc += 32) {
        v8f s0 = {}, s1 = {};
        {
            v16bf b0 = load_frag(kp, kc, D, 0,  lane);
            v16bf b1 = load_frag(kp, kc, D, 32, lane);
            s0 = bf16_wmma(qa0, b0, s0);
            s0 = bf16_wmma(qa1, b1, s0);
            b0 = load_frag(kp, kc + 16, D, 0,  lane);
            b1 = load_frag(kp, kc + 16, D, 32, lane);
            s1 = bf16_wmma(qa0, b0, s1);
            s1 = bf16_wmma(qa1, b1, s1);
        }

        float al[8];
        #pragma unroll
        for (int i = 0; i < 8; ++i) {
            const int m    = i + 8 * hh;
            const int qrow = qbase + m;
            float v0 = s0[i] * scale;
            float v1 = s1[i] * scale;
            if (kc + n      > qrow) v0 = -3.0e38f;   // causal mask
            if (kc + 16 + n > qrow) v1 = -3.0e38f;

            float mx = fmaxf(v0, v1);
            #pragma unroll
            for (int msk = 1; msk < 16; msk <<= 1) mx = fmaxf(mx, __shfl_xor(mx, msk, 32));
            const float nm = fmaxf(mrow[i], mx);
            al[i] = __expf(mrow[i] - nm);
            mrow[i] = nm;

            v0 = __expf(v0 - nm);
            v1 = __expf(v1 - nm);
            float sum = v0 + v1;
            #pragma unroll
            for (int msk = 1; msk < 16; msk <<= 1) sum += __shfl_xor(sum, msk, 32);
            lrow[i] = lrow[i] * al[i] + sum;

            pstage[wid][m * 32 + n]      = f2bf(v0);
            pstage[wid][m * 32 + 16 + n] = f2bf(v1);
        }

        #pragma unroll
        for (int f = 0; f < 4; ++f)
            #pragma unroll
            for (int i = 0; i < 8; ++i) o[f][i] *= al[i];

        Frag pf;   // reload P [16x32] as A fragment from LDS (same-wave, dscnt handled by compiler)
        {
            const unsigned short* ps = &pstage[wid][(lane & 15) * 32 + (hh << 3)];
            pf.u4[0] = *(const u32x4*)(ps);
            pf.u4[1] = *(const u32x4*)(ps + 16);
        }

        #pragma unroll
        for (int f = 0; f < 4; ++f) {
            v16bf vbf = load_frag(vp, f * 16, T, kc, lane);   // vT rows = d, cols = t
            o[f] = bf16_wmma(pf.v, vbf, o[f]);
        }
    }

    #pragma unroll
    for (int f = 0; f < 4; ++f) {
        #pragma unroll
        for (int i = 0; i < 8; ++i) {
            const int r   = qbase + i + 8 * hh;
            const int col = h * D + f * 16 + n;
            yb[((size_t)b * T + r) * C + col] = f2bf(o[f][i] / lrow[i]);
        }
    }
}

// ---------------------------------------------------------------------------
// Kernel 6: output projection.  out = y @ W_proj + b_proj   (f32 out)
//   A  = y bf16 [4096, 1024],  Bt = W_proj^T bf16 [1024, 1024]
// Same double-buffered async-to-LDS B staging as the QKV GEMM.
// ---------------------------------------------------------------------------
__global__ void __launch_bounds__(256) proj_gemm_kernel(const unsigned short* __restrict__ yb,
                                                        const unsigned short* __restrict__ Bt,
                                                        const float* __restrict__ b_proj,
                                                        float* __restrict__ out) {
    __shared__ unsigned short sB[2][64 * 32];

    const int tid  = threadIdx.x;
    const int lane = tid & 31;
    const int wid  = tid >> 5;
    const int Mbase = blockIdx.x * 128 + wid * 16;
    const int Nbase = blockIdx.y * 64;
    const int row = tid >> 2;
    const int seg = (tid & 3) << 3;

    v8f acc[4] = {{}, {}, {}, {}};
    ISSUE_B_TILE(0, 0);
    for (int i = 0; i < 32; ++i) {
        const int kk = i << 5;
        WAIT_ASYNC();
        __syncthreads();
        if (i + 1 < 32) ISSUE_B_TILE((i + 1) & 1, kk + 32);

        v16bf a = load_frag(yb, Mbase, C, kk, lane);
        const unsigned short* bs = sB[i & 1];
        #pragma unroll
        for (int f = 0; f < 4; ++f)
            acc[f] = bf16_wmma(a, lds_fragB(bs, f, lane), acc[f]);
    }

    const int hh = lane >> 4;
    const int n  = lane & 15;
    #pragma unroll
    for (int f = 0; f < 4; ++f) {
        const int c = Nbase + f * 16 + n;
        const float bias = b_proj[c];
        #pragma unroll
        for (int i = 0; i < 8; ++i) {
            const int r = Mbase + i + 8 * hh;
            out[(size_t)r * C + c] = acc[f][i] + bias;
        }
    }
}

// ---------------------------------------------------------------------------
// Launch
// ---------------------------------------------------------------------------
extern "C" void kernel_launch(void* const* d_in, const int* in_sizes, int n_in,
                              void* d_out, int out_size, void* d_ws, size_t ws_size,
                              hipStream_t stream) {
    (void)in_sizes; (void)n_in; (void)out_size; (void)ws_size;

    const float* x      = (const float*)d_in[0];
    const float* W_attn = (const float*)d_in[1];
    const float* b_attn = (const float*)d_in[2];
    const float* W_proj = (const float*)d_in[3];
    const float* b_proj = (const float*)d_in[4];
    float* out = (float*)d_out;

    // workspace layout (bf16 = unsigned short), ~50 MB total
    unsigned short* xb  = (unsigned short*)d_ws;            // [4096,1024]
    unsigned short* WaT = xb  + (size_t)Bn * T * C;         // [3072,1024]
    unsigned short* WpT = WaT + (size_t)C3 * C;             // [1024,1024]
    unsigned short* qb  = WpT + (size_t)C  * C;             // [B,H,T,D]
    unsigned short* kb  = qb  + (size_t)Bn * H * T * D;     // [B,H,T,D]
    unsigned short* vT  = kb  + (size_t)Bn * H * T * D;     // [B,H,D,T]
    unsigned short* yb  = vT  + (size_t)Bn * H * T * D;     // [4096,1024]

    const int nx = Bn * T * C;                              // 4096*1024
    cast_bf16_kernel<<<nx / 256, 256, 0, stream>>>(x, xb, nx);
    transpose_cast_kernel<<<(C * C3) / 256, 256, 0, stream>>>(W_attn, WaT, C, C3);
    transpose_cast_kernel<<<(C * C)  / 256, 256, 0, stream>>>(W_proj, WpT, C, C);

    qkv_gemm_kernel<<<dim3(32, 48), 256, 0, stream>>>(xb, WaT, b_attn, qb, kb, vT);

    rope_kernel<<<(Bn * H * T * 32) / 256, 256, 0, stream>>>(qb, kb);

    attn_kernel<<<(Bn * H * (T / 16)) / 8, 256, 0, stream>>>(qb, kb, vT, yb);

    proj_gemm_kernel<<<dim3(32, 16), 256, 0, stream>>>(yb, WpT, b_proj, out);
}
